// MultiHeadSelfAttention_29205777613372
// MI455X (gfx1250) — matrix operand order repaired and checked by reading, not compiled, not run on hardware
//
#include <hip/hip_runtime.h>

typedef unsigned short u16;
typedef __attribute__((ext_vector_type(16))) __bf16 v16bf;
typedef __attribute__((ext_vector_type(8)))  __bf16 v8bf;
typedef __attribute__((ext_vector_type(8)))  float  v8f;

#define D_MODEL 1024
#define N_HEADS 16
#define D_K     64
#define BATCH   2
#define SEQ     2048
#define TOKENS  (BATCH*SEQ)   // 4096

__device__ __forceinline__ u16 f32_to_bf16(float f) {
  unsigned u = __float_as_uint(f);
  u += 0x7FFFu + ((u >> 16) & 1u);   // round-to-nearest-even
  return (u16)(u >> 16);
}

__device__ __forceinline__ v8f v8f_zero() {
  v8f r;
#pragma unroll
  for (int i = 0; i < 8; ++i) r[i] = 0.0f;
  return r;
}

// Build a 16-element bf16 fragment from two 16-byte chunks (lo: elems 0..7, hi: 8..15)
__device__ __forceinline__ v16bf frag_from(const u16* lo, const u16* hi) {
  v8bf a = *(const v8bf*)lo;
  v8bf b = *(const v8bf*)hi;
  v16bf r;
#pragma unroll
  for (int i = 0; i < 8; ++i) { r[i] = a[i]; r[i + 8] = b[i]; }
  return r;
}

// Load one k-step of fragments: 2 A tiles (rows) + 4 B tiles (cols, from B^T).
__device__ __forceinline__ void load_ab(const u16* __restrict__ A,
                                        const u16* __restrict__ Bt, int K,
                                        int m0, int n0, int lr, int half, int k,
                                        v16bf a[2], v16bf b[4]) {
#pragma unroll
  for (int i = 0; i < 2; ++i) {
    const u16* row = A + (size_t)(m0 + i * 16 + lr) * K + k;
    a[i] = frag_from(row + half * 8, row + 16 + half * 8);
  }
#pragma unroll
  for (int j = 0; j < 4; ++j) {
    const u16* col = Bt + (size_t)(n0 + j * 16 + lr) * K + k;
    b[j] = frag_from(col + half * 8, col + 16 + half * 8);
  }
}

__device__ __forceinline__ void mma_tile(v8f acc[2][4], const v16bf a[2], const v16bf b[4]) {
#pragma unroll
  for (int i = 0; i < 2; ++i)
#pragma unroll
    for (int j = 0; j < 4; ++j)
      acc[i][j] = __builtin_amdgcn_wmma_f32_16x16x32_bf16(
          false, a[i], false, b[j], (short)0, acc[i][j], false, false);
}

// Double-buffered 32x64-per-wave GEMM core (K multiple of 64).
__device__ __forceinline__ void gemm_core(const u16* __restrict__ A,
                                          const u16* __restrict__ Bt, int K,
                                          int m0, int n0, int lr, int half,
                                          v8f acc[2][4]) {
#pragma unroll
  for (int i = 0; i < 2; ++i)
#pragma unroll
    for (int j = 0; j < 4; ++j) acc[i][j] = v8f_zero();

  v16bf a0[2], b0[4], a1[2], b1[4];
  load_ab(A, Bt, K, m0, n0, lr, half, 0, a0, b0);
  for (int k = 0; k < K - 64; k += 64) {
    load_ab(A, Bt, K, m0, n0, lr, half, k + 32, a1, b1);
    mma_tile(acc, a0, b0);
    load_ab(A, Bt, K, m0, n0, lr, half, k + 64, a0, b0);
    mma_tile(acc, a1, b1);
  }
  load_ab(A, Bt, K, m0, n0, lr, half, K - 32, a1, b1);
  mma_tile(acc, a0, b0);
  mma_tile(acc, a1, b1);
}

// ---------------------------------------------------------------- prep kernels

__global__ void cast_f32_bf16(const float* __restrict__ in, u16* __restrict__ out, int n) {
  int i = blockIdx.x * blockDim.x + threadIdx.x;
  int stride = gridDim.x * blockDim.x;
  for (; i < n; i += stride) out[i] = f32_to_bf16(in[i]);
}

// in: [K][N] fp32 (row-major)  ->  out: [N][K] bf16 (transposed)
__global__ void transpose_cast(const float* __restrict__ in, u16* __restrict__ out,
                               int K, int N) {
  int i = blockIdx.x * blockDim.x + threadIdx.x;
  int total = K * N;
  int stride = gridDim.x * blockDim.x;
  for (; i < total; i += stride) {
    int n = i / K, k = i % K;
    out[i] = f32_to_bf16(in[k * N + n]);
  }
}

// ---------------------------------------------------------------- QKV GEMM
// A: x bf16 [4096][1024], Bt: W_qkv^T bf16 [3072][1024]
// Epilogue scatters Q (scaled 1/sqrt(Dk)), K as [B*H][S][Dk], V transposed [B*H][Dk][S].
// mat / h are wave-uniform (n0 is 64-aligned; 64-blocks never straddle a 1024 boundary).
__global__ void __launch_bounds__(256)
qkv_gemm(const u16* __restrict__ A, const u16* __restrict__ Bt,
         const float* __restrict__ bias,
         u16* __restrict__ Qo, u16* __restrict__ Ko, u16* __restrict__ Vt) {
  const int K = D_MODEL;
  int lane = threadIdx.x & 31, wave = threadIdx.x >> 5;
  int lr = lane & 15, half = lane >> 4;
  int wm = wave >> 1, wn = wave & 1;
  int m0 = blockIdx.y * 128 + wm * 32;
  int n0 = blockIdx.x * 128 + wn * 64;

  v8f acc[2][4];
  gemm_core(A, Bt, K, m0, n0, lr, half, acc);

  int mat = n0 >> 10;                   // wave-uniform: 0=Q, 1=K, 2=V
  int h   = (n0 >> 6) & (N_HEADS - 1);  // wave-uniform head index

  if (mat == 0) {
#pragma unroll
    for (int i = 0; i < 2; ++i)
#pragma unroll
      for (int j = 0; j < 4; ++j)
#pragma unroll
        for (int r = 0; r < 8; ++r) {
          int m = m0 + i * 16 + half * 8 + r;
          int bb = m >> 11, s = m & (SEQ - 1);
          int dk = j * 16 + lr;
          float v = (acc[i][j][r] + bias[n0 + dk]) * 0.125f;  // pre-scale by 1/sqrt(Dk)
          Qo[(((size_t)(bb * N_HEADS + h)) * SEQ + s) * D_K + dk] = f32_to_bf16(v);
        }
  } else if (mat == 1) {
#pragma unroll
    for (int i = 0; i < 2; ++i)
#pragma unroll
      for (int j = 0; j < 4; ++j)
#pragma unroll
        for (int r = 0; r < 8; ++r) {
          int m = m0 + i * 16 + half * 8 + r;
          int bb = m >> 11, s = m & (SEQ - 1);
          int dk = j * 16 + lr;
          float v = acc[i][j][r] + bias[n0 + dk];
          Ko[(((size_t)(bb * N_HEADS + h)) * SEQ + s) * D_K + dk] = f32_to_bf16(v);
        }
  } else {
#pragma unroll
    for (int i = 0; i < 2; ++i)
#pragma unroll
      for (int j = 0; j < 4; ++j)
#pragma unroll
        for (int r = 0; r < 8; ++r) {
          int m = m0 + i * 16 + half * 8 + r;
          int bb = m >> 11, s = m & (SEQ - 1);
          int dk = j * 16 + lr;
          float v = acc[i][j][r] + bias[n0 + dk];
          Vt[(((size_t)(bb * N_HEADS + h)) * D_K + dk) * SEQ + s] = f32_to_bf16(v);
        }
  }
}

// ---------------------------------------------------------------- attention
// One wave = 16 queries; stream 128-key chunks; online softmax; out -> bf16 [B,S,D].
__global__ void __launch_bounds__(256)
attention(const u16* __restrict__ Q, const u16* __restrict__ Kc,
          const u16* __restrict__ Vt, u16* __restrict__ AttnOut) {
  __shared__ u16 lds_p[8][16][128];   // per-wave P tile (bf16), 32 KB
  int lane = threadIdx.x & 31, wave = threadIdx.x >> 5;
  int lr = lane & 15, half = lane >> 4;
  int q0 = blockIdx.x * 128 + wave * 16;
  int h = blockIdx.y, bz = blockIdx.z;
  int bh = bz * N_HEADS + h;
  const u16* Qb = Q  + (size_t)bh * SEQ * D_K;
  const u16* Kb = Kc + (size_t)bh * SEQ * D_K;
  const u16* Vb = Vt + (size_t)bh * D_K * SEQ;

  // Q fragments are loop invariant (Dk = 64 -> 2 k-steps of 32)
  v16bf aq[2];
#pragma unroll
  for (int t = 0; t < 2; ++t) {
    const u16* row = Qb + (size_t)(q0 + lr) * D_K + t * 32;
    aq[t] = frag_from(row + half * 8, row + 16 + half * 8);
  }

  v8f oacc[4];
#pragma unroll
  for (int d = 0; d < 4; ++d) oacc[d] = v8f_zero();
  float row_m[8], row_l[8];
#pragma unroll
  for (int r = 0; r < 8; ++r) { row_m[r] = -1e30f; row_l[r] = 0.0f; }

  for (int kc = 0; kc < SEQ; kc += 128) {
    // scores: 16 x 128 (Q pre-scaled by 1/sqrt(Dk)); loop unrolled -> loads hoist freely
    v8f sc[8];
#pragma unroll
    for (int j = 0; j < 8; ++j) {
      const u16* col = Kb + (size_t)(kc + j * 16 + lr) * D_K;
      v16bf b0 = frag_from(col + half * 8, col + 16 + half * 8);
      v16bf b1 = frag_from(col + 32 + half * 8, col + 48 + half * 8);
      v8f z = v8f_zero();
      z = __builtin_amdgcn_wmma_f32_16x16x32_bf16(false, aq[0], false, b0, (short)0, z, false, false);
      sc[j] = __builtin_amdgcn_wmma_f32_16x16x32_bf16(false, aq[1], false, b1, (short)0, z, false, false);
    }

    // online softmax (row = half*8 + r, 16 lanes per half hold one row)
#pragma unroll
    for (int r = 0; r < 8; ++r) {
      float mx = sc[0][r];
#pragma unroll
      for (int j = 1; j < 8; ++j) mx = fmaxf(mx, sc[j][r]);
      mx = fmaxf(mx, __shfl_xor(mx, 1, 32));
      mx = fmaxf(mx, __shfl_xor(mx, 2, 32));
      mx = fmaxf(mx, __shfl_xor(mx, 4, 32));
      mx = fmaxf(mx, __shfl_xor(mx, 8, 32));
      float m_new = fmaxf(row_m[r], mx);
      float corr = __expf(row_m[r] - m_new);
      float sum = 0.0f;
#pragma unroll
      for (int j = 0; j < 8; ++j) {
        float p = __expf(sc[j][r] - m_new);
        sc[j][r] = p;
        sum += p;
      }
      sum += __shfl_xor(sum, 1, 32);
      sum += __shfl_xor(sum, 2, 32);
      sum += __shfl_xor(sum, 4, 32);
      sum += __shfl_xor(sum, 8, 32);
      row_l[r] = row_l[r] * corr + sum;
      row_m[r] = m_new;
#pragma unroll
      for (int d = 0; d < 4; ++d) oacc[d][r] = oacc[d][r] * corr;
    }

    // transpose P through per-wave LDS (C-layout -> A-layout); wave-private,
    // DS ops are in-order within a wave so no barrier needed
#pragma unroll
    for (int j = 0; j < 8; ++j)
#pragma unroll
      for (int r = 0; r < 8; ++r)
        lds_p[wave][half * 8 + r][j * 16 + lr] = f32_to_bf16(sc[j][r]);

    // P(16x128) @ V(128x64): A-frags from LDS, B-frags straight from global V^T
#pragma unroll
    for (int t = 0; t < 4; ++t) {
      const u16* prow = &lds_p[wave][lr][t * 32];
      v16bf ap = frag_from(prow + half * 8, prow + 16 + half * 8);
#pragma unroll
      for (int d = 0; d < 4; ++d) {
        const u16* vcol = Vb + (size_t)(d * 16 + lr) * SEQ + kc + t * 32;
        v16bf bv = frag_from(vcol + half * 8, vcol + 16 + half * 8);
        oacc[d] = __builtin_amdgcn_wmma_f32_16x16x32_bf16(
            false, ap, false, bv, (short)0, oacc[d], false, false);
      }
    }
  }

  // normalize + write back into [B, S, D] bf16 for the output projection
#pragma unroll
  for (int r = 0; r < 8; ++r) {
    float inv = 1.0f / row_l[r];
    int s = q0 + half * 8 + r;
    size_t rowoff = ((size_t)bz * SEQ + s) * D_MODEL + (size_t)h * D_K;
#pragma unroll
    for (int d = 0; d < 4; ++d)
      AttnOut[rowoff + d * 16 + lr] = f32_to_bf16(oacc[d][r] * inv);
  }
}

// ---------------------------------------------------------------- out GEMM
__global__ void __launch_bounds__(256)
out_gemm(const u16* __restrict__ A, const u16* __restrict__ Bt,
         const float* __restrict__ bias, float* __restrict__ Out) {
  const int K = D_MODEL;
  int lane = threadIdx.x & 31, wave = threadIdx.x >> 5;
  int lr = lane & 15, half = lane >> 4;
  int wm = wave >> 1, wn = wave & 1;
  int m0 = blockIdx.y * 128 + wm * 32;
  int n0 = blockIdx.x * 128 + wn * 64;

  v8f acc[2][4];
  gemm_core(A, Bt, K, m0, n0, lr, half, acc);

#pragma unroll
  for (int i = 0; i < 2; ++i)
#pragma unroll
    for (int j = 0; j < 4; ++j)
#pragma unroll
      for (int r = 0; r < 8; ++r) {
        int m = m0 + i * 16 + half * 8 + r;
        int c = n0 + j * 16 + lr;
        Out[(size_t)m * D_MODEL + c] = acc[i][j][r] + bias[c];
      }
}

// ---------------------------------------------------------------- launch

extern "C" void kernel_launch(void* const* d_in, const int* in_sizes, int n_in,
                              void* d_out, int out_size, void* d_ws, size_t ws_size,
                              hipStream_t stream) {
  (void)in_sizes; (void)n_in; (void)out_size; (void)ws_size;
  const float* x     = (const float*)d_in[0];
  const float* W_qkv = (const float*)d_in[1];
  const float* b_qkv = (const float*)d_in[2];
  const float* W_out = (const float*)d_in[3];
  const float* b_out = (const float*)d_in[4];
  float* out = (float*)d_out;

  char* ws = (char*)d_ws;
  u16* xb    = (u16*)(ws);                 //  8 MB : x bf16 [4096][1024]
  u16* wqkvT = (u16*)(ws + 8388608);       //  6 MB : W_qkv^T bf16 [3072][1024]
  u16* woutT = (u16*)(ws + 14680064);      //  2 MB : W_out^T bf16 [1024][1024]
  u16* Qb    = (u16*)(ws + 16777216);      //  8 MB : [B*H][S][Dk]
  u16* Kb    = (u16*)(ws + 25165824);      //  8 MB : [B*H][S][Dk]
  u16* Vt    = (u16*)(ws + 33554432);      //  8 MB : [B*H][Dk][S]
  u16* attnb = (u16*)(ws + 41943040);      //  8 MB : attn out bf16 [4096][1024]

  cast_f32_bf16<<<dim3(1024), dim3(256), 0, stream>>>(x, xb, TOKENS * D_MODEL);
  transpose_cast<<<dim3(1024), dim3(256), 0, stream>>>(W_qkv, wqkvT, D_MODEL, 3 * D_MODEL);
  transpose_cast<<<dim3(512),  dim3(256), 0, stream>>>(W_out, woutT, D_MODEL, D_MODEL);

  qkv_gemm<<<dim3(24, 32), dim3(256), 0, stream>>>(xb, wqkvT, b_qkv, Qb, Kb, Vt);
  attention<<<dim3(SEQ / 128, N_HEADS, BATCH), dim3(256), 0, stream>>>(Qb, Kb, Vt, attnb);
  out_gemm<<<dim3(8, 32), dim3(256), 0, stream>>>(attnb, woutT, b_out, out);
}
